// RoIAlignHead_29540785062163
// MI455X (gfx1250) — matrix operand order repaired and checked by reading, not compiled, not run on hardware
//
#include <hip/hip_runtime.h>
#include <hip/hip_bf16.h>

// ---------------------------------------------------------------------------
// RoIAlign + dual linear head (cls 80, reg 320), bf16 WMMA GEMM for gfx1250.
//
//   B=4, C=256, H=W=50, N=1024 rois, P=7, GRID=2, K = C*P*P = 12544
//   out: cls (1024x80) ++ reg (1024x320), fp32
//
// GEMM uses v_wmma_f32_16x16x32_bf16 with double-buffered async global->LDS
// staging (ASYNCcnt pipeline) when the gfx1250 async builtins are available.
// ---------------------------------------------------------------------------

typedef __attribute__((ext_vector_type(16))) __bf16 v16bf;
typedef __attribute__((ext_vector_type(8)))  float  v8f;
typedef __attribute__((ext_vector_type(4)))  int    v4i;

struct U4x2 { uint4 x, y; };   // 32 bytes -> bit_cast to v16bf

#define ROI_N     1024
#define ROI_C     256
#define ROI_H     50
#define ROI_W     50
#define ROI_K     12544        // 256 * 49
#define N_CLS     80
#define N_REG     320
#define N_TOT     400          // 80 + 320

// ---- CDNA5 async global->LDS support probes -------------------------------
#if defined(__has_builtin)
#if __has_builtin(__builtin_amdgcn_global_load_async_to_lds_b128)
#define HAVE_ASYNC_LDS 1
#endif
#if __has_builtin(__builtin_amdgcn_s_wait_asynccnt)
#define WAIT_ASYNC(n) __builtin_amdgcn_s_wait_asynccnt(n)
#else
#define WAIT_ASYNC(n) asm volatile("s_wait_asynccnt %0" :: "n"(n) : "memory")
#endif
#endif

#if defined(HAVE_ASYNC_LDS)
// builtin signature: (v4i addrspace(1)* src, v4i addrspace(3)* dst, Ii offset, Ii cpol)
typedef __attribute__((address_space(1))) v4i gv4i_t;
typedef __attribute__((address_space(3))) v4i lv4i_t;

__device__ __forceinline__ void async_b128(const void* g, const void* l) {
    // generic LDS pointers carry the LDS byte offset in the low 32 bits
    __builtin_amdgcn_global_load_async_to_lds_b128(
        (gv4i_t*)(unsigned long long)g,
        (lv4i_t*)(unsigned int)(unsigned long long)l,
        0, 0);
}
#endif

// ---------------------------------------------------------------------------
// Kernel 1: concatenate + convert weights (w_cls ++ w_reg) to bf16, row-major
// [400][12544].  400*12544 = 5,017,600 = 19600 blocks * 256 threads.
// ---------------------------------------------------------------------------
__global__ void wconv_kernel(const float* __restrict__ w_cls,
                             const float* __restrict__ w_reg,
                             __bf16* __restrict__ wB)
{
    size_t i = (size_t)blockIdx.x * 256 + threadIdx.x;   // < 400*12544
    size_t j = i / ROI_K;
    size_t k = i - j * ROI_K;
    float v = (j < N_CLS) ? w_cls[j * ROI_K + k]
                          : w_reg[(j - N_CLS) * ROI_K + k];
    wB[i] = (__bf16)v;
}

// ---------------------------------------------------------------------------
// Kernel 2: RoIAlign.  One block (256 threads) per ROI.  The 14 y-samples and
// 14 x-samples are computed once into LDS and reused by all 256 channels.
// Output bf16, layout flat[n][c*49 + py*7 + px] (contiguous stores).
// Features (10 MB) are fully L2-resident, so the bilinear gathers are cheap.
// ---------------------------------------------------------------------------
__global__ void __launch_bounds__(256) roialign_kernel(
        const float* __restrict__ feat,
        const float* __restrict__ rois,
        const int*   __restrict__ ridx,
        __bf16*      __restrict__ flat)
{
    const int n   = blockIdx.x;
    const int tid = threadIdx.x;

    __shared__ int   s_yl[14], s_yh[14], s_xl[14], s_xh[14];
    __shared__ float s_ly[14], s_hy[14], s_lx[14], s_hx[14];
    __shared__ int   s_vy[14], s_vx[14];
    __shared__ int   s_b;

    if (tid < 28) {
        const int  i   = tid % 14;
        const bool isx = tid >= 14;
        const float scale = 0.0675f;
        float lo = rois[n * 4 + (isx ? 0 : 1)] * scale - 0.5f;
        float hi = rois[n * 4 + (isx ? 2 : 3)] * scale - 0.5f;
        float binsz = (hi - lo) * (1.0f / 7.0f);
        float off   = (float)(i >> 1) + ((float)(i & 1) + 0.5f) * 0.5f;
        float coord = lo + off * binsz;
        const int limit = isx ? ROI_W : ROI_H;   // both 50
        int valid = (coord > -1.0f) && (coord < (float)limit);
        float c = fmaxf(coord, 0.0f);
        int low  = min((int)floorf(c), limit - 1);
        int high = min(low + 1, limit - 1);
        if (low >= limit - 1) c = (float)low;
        float l = c - (float)low;
        if (isx) { s_xl[i] = low; s_xh[i] = high; s_lx[i] = l; s_hx[i] = 1.0f - l; s_vx[i] = valid; }
        else     { s_yl[i] = low; s_yh[i] = high; s_ly[i] = l; s_hy[i] = 1.0f - l; s_vy[i] = valid; }
    }
    if (tid == 0) s_b = ridx[n];
    __syncthreads();

    const float* fb = feat + (size_t)s_b * (ROI_C * ROI_H * ROI_W);

    for (int it = 0; it < 49; ++it) {
        int o   = it * 256 + tid;          // 0..12543
        int c   = o / 49;
        int bin = o - c * 49;
        int py  = bin / 7;
        int px  = bin - py * 7;
        const float* fc = fb + (size_t)c * (ROI_H * ROI_W);

        float acc = 0.0f;
        #pragma unroll
        for (int gy = 0; gy < 2; ++gy) {
            int yi = py * 2 + gy;
            if (!s_vy[yi]) continue;
            const float* rl = fc + s_yl[yi] * ROI_W;
            const float* rh = fc + s_yh[yi] * ROI_W;
            float ly = s_ly[yi], hy = s_hy[yi];
            #pragma unroll
            for (int gx = 0; gx < 2; ++gx) {
                int xi = px * 2 + gx;
                if (!s_vx[xi]) continue;
                int xl = s_xl[xi], xh = s_xh[xi];
                float lx = s_lx[xi], hx = s_hx[xi];
                acc += hy * (hx * rl[xl] + lx * rl[xh])
                     + ly * (hx * rh[xl] + lx * rh[xh]);
            }
        }
        flat[(size_t)n * ROI_K + o] = (__bf16)(acc * 0.25f);
    }
}

// ---------------------------------------------------------------------------
// Kernel 3: GEMM  out[M=1024, N=400] = flat[1024,12544](bf16) x W^T + bias.
// Block = 256 threads = 8 waves -> 32x64 output tile (2 M-tiles x 4 N-tiles).
// K-step = 64: per buffer stage A 32x64 (4 KB) and B 64x64 (8 KB) in LDS,
// two v_wmma_f32_16x16x32_bf16 per wave per step, fp32 accumulation.
// Double-buffered async global->LDS pipeline (ASYNCcnt) when available.
// Columns 0..79 -> cls output (+b_cls); 80..399 -> reg output (+b_reg).
// Weight rows >= 400 are clamped to row 0; their columns are never stored.
// ---------------------------------------------------------------------------
__global__ void __launch_bounds__(256, 2) gemm_kernel(
        const __bf16* __restrict__ A,     // [1024][12544] bf16
        const __bf16* __restrict__ Bw,    // [400][12544]  bf16
        const float*  __restrict__ b_cls,
        const float*  __restrict__ b_reg,
        float*        __restrict__ out)   // cls[1024*80] ++ reg[1024*320]
{
    __shared__ __bf16 sA[2][32 * 64];     //  8 KB
    __shared__ __bf16 sB[2][64 * 64];     // 16 KB

    const int t    = threadIdx.x;
    const int m0   = blockIdx.x * 32;
    const int n0   = blockIdx.y * 64;
    const int w    = t >> 5;
    const int lane = t & 31;
    const int mt   = w >> 2;              // 0..1
    const int nt   = w & 3;               // 0..3
    const int r    = lane & 15;
    const int hiH  = (lane & 16) ? 1 : 0;
    const int kbA  = hiH ? 8  : 0;        // A: K pairs {0-7,16-23} / {8-15,24-31}
    const int kbB  = hiH ? 8  : 0;        // B: K pairs {0-7,16-23} / {8-15,24-31}

    // cooperative staging indices (K-step 64)
    const int arow = t >> 3, achk = (t & 7) * 8;    // one b128 (8 bf16) per thread
    const int brow = t >> 2, bchk = (t & 3) * 16;   // two b128 (16 bf16) per thread
    const int browC = (n0 + brow) < N_TOT ? (n0 + brow) : 0;  // clamp; cols >=400 discarded
    const __bf16* gA = A  + (size_t)(m0 + arow) * ROI_K + achk;
    const __bf16* gB = Bw + (size_t)browC * ROI_K + bchk;

    __bf16* lA = &sA[0][arow * 64 + achk];
    __bf16* lB = &sB[0][brow * 64 + bchk];
    const int bufStride  = 32 * 64;       // sA buffer stride (elements)
    const int bufStrideB = 64 * 64;       // sB buffer stride (elements)

    v8f acc = {};

#if defined(HAVE_ASYNC_LDS)
    // -------- pipelined path: async global->LDS, double-buffered ----------
    // prologue: stage tile 0 into buffer 0 (3 async b128 per thread)
    async_b128(gA,      lA);
    async_b128(gB,      lB);
    async_b128(gB + 8,  lB + 8);

    for (int kt = 0, i = 0; kt < ROI_K; kt += 64, ++i) {
        const int buf = i & 1;
        const bool has_next = (kt + 64) < ROI_K;

        __syncthreads();                  // all waves done reading buf^1
        if (has_next) {
            const int nb = (buf ^ 1);
            async_b128(gA + kt + 64,     lA + nb * bufStride);
            async_b128(gB + kt + 64,     lB + nb * bufStrideB);
            async_b128(gB + kt + 64 + 8, lB + nb * bufStrideB + 8);
            WAIT_ASYNC(3);                // drain current buf, keep next in flight
        } else {
            WAIT_ASYNC(0);
        }
        __syncthreads();                  // buf's data visible to all waves

        #pragma unroll
        for (int s = 0; s < 2; ++s) {
            const int kofs = s * 32;
            U4x2 ta, tb;
            const __bf16* pa = &sA[buf][(mt * 16 + r) * 64 + kofs + kbA];
            ta.x = *(const uint4*)(pa);
            ta.y = *(const uint4*)(pa + 16);
            const __bf16* pb = &sB[buf][(nt * 16 + r) * 64 + kofs + kbB];
            tb.x = *(const uint4*)(pb);
            tb.y = *(const uint4*)(pb + 16);
            v16bf av = __builtin_bit_cast(v16bf, ta);
            v16bf bv = __builtin_bit_cast(v16bf, tb);
            acc = __builtin_amdgcn_wmma_f32_16x16x32_bf16(
                      false, av, false, bv, (short)0, acc, false, false);
        }
    }
#else
    // -------- fallback path: synchronous VGPR staging ----------------------
    for (int kt = 0; kt < ROI_K; kt += 64) {
        __syncthreads();
        *(uint4*)lA       = *(const uint4*)(gA + kt);
        *(uint4*)lB       = *(const uint4*)(gB + kt);
        *(uint4*)(lB + 8) = *(const uint4*)(gB + kt + 8);
        if (kt + 64 < ROI_K) {
            __builtin_prefetch(gA + kt + 64, 0, 0);
            __builtin_prefetch(gB + kt + 64, 0, 0);
        }
        __syncthreads();
        #pragma unroll
        for (int s = 0; s < 2; ++s) {
            const int kofs = s * 32;
            U4x2 ta, tb;
            const __bf16* pa = &sA[0][(mt * 16 + r) * 64 + kofs + kbA];
            ta.x = *(const uint4*)(pa);
            ta.y = *(const uint4*)(pa + 16);
            const __bf16* pb = &sB[0][(nt * 16 + r) * 64 + kofs + kbB];
            tb.x = *(const uint4*)(pb);
            tb.y = *(const uint4*)(pb + 16);
            v16bf av = __builtin_bit_cast(v16bf, ta);
            v16bf bv = __builtin_bit_cast(v16bf, tb);
            acc = __builtin_amdgcn_wmma_f32_16x16x32_bf16(
                      false, av, false, bv, (short)0, acc, false, false);
        }
    }
#endif

    // Store: VGPR i holds M = i + 8*hiH, N = lane&15 (ISA 7.12.2 C/D layout)
    const int gn = n0 + nt * 16 + r;
    if (gn >= N_TOT) return;
    const int mbase = m0 + mt * 16 + hiH * 8;

    float  bias;
    float* op;
    int    ldo;
    if (gn < N_CLS) { bias = b_cls[gn];         op = out + gn;                                   ldo = N_CLS; }
    else            { bias = b_reg[gn - N_CLS]; op = out + (size_t)ROI_N * N_CLS + (gn - N_CLS); ldo = N_REG; }

    #pragma unroll
    for (int i = 0; i < 8; ++i)
        op[(size_t)(mbase + i) * ldo] = acc[i] + bias;
}

// ---------------------------------------------------------------------------
// Launch: inputs in setup_inputs() order:
//   0 features, 1 rois, 2 roi_indices, 3 w_cls, 4 b_cls, 5 w_reg, 6 b_reg
// ---------------------------------------------------------------------------
extern "C" void kernel_launch(void* const* d_in, const int* in_sizes, int n_in,
                              void* d_out, int out_size, void* d_ws, size_t ws_size,
                              hipStream_t stream)
{
    const float* feat  = (const float*)d_in[0];
    const float* rois  = (const float*)d_in[1];
    const int*   ridx  = (const int*)  d_in[2];
    const float* w_cls = (const float*)d_in[3];
    const float* b_cls = (const float*)d_in[4];
    const float* w_reg = (const float*)d_in[5];
    const float* b_reg = (const float*)d_in[6];
    float* out = (float*)d_out;

    // workspace: flat bf16 [1024][12544] then weights bf16 [400][12544]
    __bf16* flatB = (__bf16*)d_ws;
    __bf16* wB    = flatB + (size_t)ROI_N * ROI_K;

    wconv_kernel<<<19600, 256, 0, stream>>>(w_cls, w_reg, wB);
    roialign_kernel<<<ROI_N, 256, 0, stream>>>(feat, rois, ridx, flatB);
    gemm_kernel<<<dim3(ROI_N / 32, (N_TOT + 63) / 64), 256, 0, stream>>>(
        flatB, wB, b_cls, b_reg, out);
}
